// Attention_40080634806517
// MI455X (gfx1250) — matrix operand order repaired and checked by reading, not compiled, not run on hardware
//
#include <hip/hip_runtime.h>

typedef __attribute__((ext_vector_type(16))) __bf16 v16bf;
typedef __attribute__((ext_vector_type(8)))  __bf16 bf16x8;
typedef __attribute__((ext_vector_type(4)))  __bf16 bf16x4;
typedef __attribute__((ext_vector_type(8)))  float  v8f;
typedef __attribute__((ext_vector_type(4)))  float  f32x4;
typedef __attribute__((ext_vector_type(4)))  unsigned int u32x4;
typedef __attribute__((ext_vector_type(8)))  int    i32x8;
typedef __attribute__((ext_vector_type(4)))  int    i32x4;

// Problem constants (match reference)
constexpr int CB    = 2;
constexpr int CS    = 2048;
constexpr int CD    = 4096;
constexpr int CNH   = 32;
constexpr int CNKV  = 8;
constexpr int CHD   = 128;
constexpr int CNREP = CNH / CNKV;   // 4
constexpr int CM    = CB * CS;      // 4096 token rows

#define WMMA_BF16(a, b, c) \
  __builtin_amdgcn_wmma_f32_16x16x32_bf16(false, (a), false, (b), (short)0, (c), false, false)

static __device__ __forceinline__ v8f zero8() {
  v8f z = {0.f, 0.f, 0.f, 0.f, 0.f, 0.f, 0.f, 0.f};
  return z;
}

// A fragment (16x32, M x K): lane holds row = lane&15; elems 0..7 -> K=half*8+i,
// elems 8..15 -> K=16+half*8+(i-8)
static __device__ __forceinline__ v16bf load_a_frag(const __bf16* rowptr, int half) {
  bf16x8 lo = *(const bf16x8*)(rowptr + half * 8);
  bf16x8 hi = *(const bf16x8*)(rowptr + 16 + half * 8);
  return __builtin_shufflevector(lo, hi, 0,1,2,3,4,5,6,7,8,9,10,11,12,13,14,15);
}

// B fragment (32x16, K x N) read from B^T rows: lane holds col N = lane&15,
// elems 0..7 -> K=half*8+i, elems 8..15 -> K=16+half*8+(i-8)
static __device__ __forceinline__ v16bf load_b_frag(const __bf16* rowptr, int half) {
  bf16x8 lo = *(const bf16x8*)(rowptr + half * 8);
  bf16x8 hi = *(const bf16x8*)(rowptr + 16 + half * 8);
  return __builtin_shufflevector(lo, hi, 0,1,2,3,4,5,6,7,8,9,10,11,12,13,14,15);
}

// ---------------------------------------------------------------------------
// Tensor Data Mover: 2-D bf16 tile load, global -> LDS (with LDS row padding)
// D# packing per CDNA5 ISA.
// ---------------------------------------------------------------------------
#define HAS_TDM __has_builtin(__builtin_amdgcn_tensor_load_to_lds)

#if HAS_TDM
static __device__ __forceinline__ void tdm_load_tile_bf16(
    unsigned int lds_byte_addr, const void* gptr,
    unsigned int tensor_d0, unsigned int tensor_d1,
    unsigned int tile_d0, unsigned int tile_d1,
    unsigned long long stride0_elems,
    unsigned int pad_interval_code, unsigned int pad_amount_code, bool pad_en) {
  unsigned long long ga = (unsigned long long)(uintptr_t)gptr;
  u32x4 g0;
  g0[0] = 1u;                                       // count=1, user descriptor
  g0[1] = lds_byte_addr;                            // lds_addr [63:32]
  g0[2] = (unsigned int)ga;                         // global_addr[31:0]
  g0[3] = (unsigned int)((ga >> 32) & 0x01FFFFFFu)  // global_addr[56:32]
          | 0x80000000u;                            // type=2 ("image") [127:126]
  i32x8 g1;
  unsigned int w0 = (1u << 16)                      // data_size=1 -> 2 bytes
                  | (pad_en ? (1u << 20) : 0u)      // pad_enable
                  | (pad_interval_code << 22)       // pad_interval [24:22]
                  | (pad_amount_code << 25);        // pad_amount   [31:25]
  g1[0] = (int)w0;                                  // wg_mask=0 (not in cluster)
  g1[1] = (int)((tensor_d0 & 0xFFFFu) << 16);       // tensor_dim0[15:0] @ [63:48]
  g1[2] = (int)((tensor_d0 >> 16) | ((tensor_d1 & 0xFFFFu) << 16));
  g1[3] = (int)((tensor_d1 >> 16) | (tile_d0 << 16));
  g1[4] = (int)(tile_d1 & 0xFFFFu);                 // tile_dim1; tile_dim2=0
  g1[5] = (int)(unsigned int)(stride0_elems & 0xFFFFFFFFu);
  g1[6] = (int)(unsigned int)((stride0_elems >> 32) & 0xFFFFu);  // stride0[47:32]
  g1[7] = 0;                                        // tensor_dim1_stride=0 (2D)
  i32x4 gz = {0, 0, 0, 0};
#if defined(__clang_major__) && (__clang_major__ >= 23)
  i32x8 gz8 = {0, 0, 0, 0, 0, 0, 0, 0};
  __builtin_amdgcn_tensor_load_to_lds(g0, g1, gz, gz, gz8, 0);
#else
  __builtin_amdgcn_tensor_load_to_lds(g0, g1, gz, gz, 0);
#endif
}
#endif

// ---------------------------------------------------------------------------
// f32 -> bf16 straight conversion (vectorized x4)
// ---------------------------------------------------------------------------
__global__ __launch_bounds__(256)
void cvt_bf16_kernel(const float* __restrict__ in, __bf16* __restrict__ out, int n4) {
  int i = blockIdx.x * 256 + threadIdx.x;
  if (i < n4) {
    f32x4 v = ((const f32x4*)in)[i];
    bf16x4 o;
    o[0] = (__bf16)v[0]; o[1] = (__bf16)v[1];
    o[2] = (__bf16)v[2]; o[3] = (__bf16)v[3];
    ((bf16x4*)out)[i] = o;
  }
}

// ---------------------------------------------------------------------------
// f32 [K][N] -> bf16 [N][K] transpose-convert (32x32 LDS tiles)
// ---------------------------------------------------------------------------
__global__ __launch_bounds__(256)
void cvt_transpose_kernel(const float* __restrict__ W, __bf16* __restrict__ Wt,
                          int K, int N) {
  __shared__ float tile[32][33];
  int n0 = blockIdx.x * 32;
  int k0 = blockIdx.y * 32;
  int tx = threadIdx.x & 31;
  int ty = threadIdx.x >> 5;   // 0..7
#pragma unroll
  for (int i = 0; i < 4; ++i) {
    int r = ty + i * 8;
    tile[r][tx] = W[(size_t)(k0 + r) * N + (n0 + tx)];
  }
  __syncthreads();
#pragma unroll
  for (int i = 0; i < 4; ++i) {
    int r = ty + i * 8;
    Wt[(size_t)(n0 + r) * K + (k0 + tx)] = (__bf16)tile[tx][r];
  }
}

// ---------------------------------------------------------------------------
// GEMM: C[M][N] = A[M][K] (bf16) * Bt[N][K]^T (bf16), f32 accumulate.
// Block 128x128x32, 8 waves (2x4), double-buffered LDS for load/WMMA overlap.
// ---------------------------------------------------------------------------
template <typename OutT>
__global__ __launch_bounds__(256)
void gemm_bf16_kernel(const __bf16* __restrict__ A, const __bf16* __restrict__ Bt,
                      OutT* __restrict__ C, int M, int N, int K) {
  constexpr int BM = 128, BN = 128, BK = 32, PAD = 8;
  __shared__ __bf16 As[2][BM][BK + PAD];
  __shared__ __bf16 Bs[2][BN][BK + PAD];

  const int tid  = threadIdx.x;
  const int lane = tid & 31;
  const int wid  = tid >> 5;
  const int half = lane >> 4;
  const int l16  = lane & 15;
  const int wm   = wid >> 2;   // 0..1
  const int wn   = wid & 3;    // 0..3
  const int m0   = blockIdx.y * BM;
  const int n0   = blockIdx.x * BN;

  // Per-thread staging coordinates: 512 chunks of 8 bf16 per tile, 2/thread.
  const int r0 = tid >> 2;            // 0..63
  const int c0 = (tid & 3) * 8;
  const int r1 = r0 + 64;             // 64..127
  const int c1 = c0;

  const __bf16* Abase = A + (size_t)m0 * K;
  const __bf16* Bbase = Bt + (size_t)n0 * K;

  // Preload tile 0
  *(bf16x8*)&As[0][r0][c0] = *(const bf16x8*)&Abase[(size_t)r0 * K + c0];
  *(bf16x8*)&As[0][r1][c1] = *(const bf16x8*)&Abase[(size_t)r1 * K + c1];
  *(bf16x8*)&Bs[0][r0][c0] = *(const bf16x8*)&Bbase[(size_t)r0 * K + c0];
  *(bf16x8*)&Bs[0][r1][c1] = *(const bf16x8*)&Bbase[(size_t)r1 * K + c1];
  __syncthreads();

  v8f acc[4][2];
#pragma unroll
  for (int mi = 0; mi < 4; ++mi)
#pragma unroll
    for (int ni = 0; ni < 2; ++ni) acc[mi][ni] = zero8();

  const int nt = K / BK;
  int cur = 0;
  for (int kt = 0; kt < nt; ++kt) {
    const bool more = (kt + 1) < nt;
    bf16x8 ra0, ra1, rb0, rb1;
    if (more) {  // uniform branch: issue next-tile global loads early
      int kk = (kt + 1) * BK;
      ra0 = *(const bf16x8*)&Abase[(size_t)r0 * K + kk + c0];
      ra1 = *(const bf16x8*)&Abase[(size_t)r1 * K + kk + c1];
      rb0 = *(const bf16x8*)&Bbase[(size_t)r0 * K + kk + c0];
      rb1 = *(const bf16x8*)&Bbase[(size_t)r1 * K + kk + c1];
    }

    v16bf af[4], bfr[2];
#pragma unroll
    for (int mi = 0; mi < 4; ++mi)
      af[mi] = load_a_frag(&As[cur][wm * 64 + mi * 16 + l16][0], half);
#pragma unroll
    for (int ni = 0; ni < 2; ++ni)
      bfr[ni] = load_b_frag(&Bs[cur][wn * 32 + ni * 16 + l16][0], half);

#pragma unroll
    for (int mi = 0; mi < 4; ++mi)
#pragma unroll
      for (int ni = 0; ni < 2; ++ni)
        acc[mi][ni] = WMMA_BF16(af[mi], bfr[ni], acc[mi][ni]);

    if (more) {  // commit next tile while this tile's WMMAs drain
      int nxt = cur ^ 1;
      *(bf16x8*)&As[nxt][r0][c0] = ra0;
      *(bf16x8*)&As[nxt][r1][c1] = ra1;
      *(bf16x8*)&Bs[nxt][r0][c0] = rb0;
      *(bf16x8*)&Bs[nxt][r1][c1] = rb1;
    }
    __syncthreads();
    cur ^= 1;
  }

  // C/D layout: reg j -> row = j + 8*half, col = lane&15
#pragma unroll
  for (int mi = 0; mi < 4; ++mi)
#pragma unroll
    for (int ni = 0; ni < 2; ++ni)
#pragma unroll
      for (int j = 0; j < 8; ++j) {
        int r = m0 + wm * 64 + mi * 16 + j + 8 * half;
        int c = n0 + wn * 32 + ni * 16 + l16;
        C[(size_t)r * N + c] = (OutT)acc[mi][ni][j];
      }
}

// ---------------------------------------------------------------------------
// RoPE on bf16 tensor laid out [B*S][nheads*HD]
// ---------------------------------------------------------------------------
__global__ __launch_bounds__(256)
void rope_kernel(__bf16* __restrict__ t, const float* __restrict__ fc,
                 const float* __restrict__ fs, int nheads) {
  const int row  = blockIdx.x;            // 0 .. B*S-1
  const int srow = row & (CS - 1);
  const int ld   = nheads * CHD;
  for (int p = threadIdx.x; p < nheads * (CHD / 2); p += 256) {
    int h = p >> 6;        // HD/2 == 64
    int d = p & 63;
    size_t base = (size_t)row * ld + h * CHD + 2 * d;
    float c = fc[srow * 64 + d];
    float s = fs[srow * 64 + d];
    float re = (float)t[base];
    float im = (float)t[base + 1];
    t[base]     = (__bf16)(re * c - im * s);
    t[base + 1] = (__bf16)(re * s + im * c);
  }
}

// ---------------------------------------------------------------------------
// Flash attention (causal, GQA). 128 query rows/block, 8 waves x 16 rows.
// K tile streamed into LDS by the Tensor Data Mover (wave 0), overlapped with
// the all-thread V transpose staging; Q fragments stay in registers.
// ---------------------------------------------------------------------------
__global__ __launch_bounds__(256)
void attn_kernel(const __bf16* __restrict__ q, const __bf16* __restrict__ k,
                 const __bf16* __restrict__ v, __bf16* __restrict__ o) {
  constexpr int BM = 128, BN = 64, PAD = 8;
  __shared__ __bf16 Ks[BN][CHD + PAD];       // 64 x 136  (keys row-major)
  __shared__ __bf16 Vt[CHD][BN + PAD];       // 128 x 72  (V transposed)
  __shared__ __bf16 Ps[8][16][BN + PAD];     // per-wave P staging

  const int tid  = threadIdx.x;
  const int lane = tid & 31;
  const int wid  = tid >> 5;
  const int half = lane >> 4;
  const int l16  = lane & 15;

  const int qb = blockIdx.x * BM;
  const int h  = blockIdx.y;
  const int g  = h / CNREP;
  const int b  = blockIdx.z;

  const int ldq  = CNH * CHD;
  const int ldkv = CNKV * CHD;
  const __bf16* qhp = q + (size_t)(b * CS) * ldq + (size_t)h * CHD;
  const __bf16* khp = k + (size_t)(b * CS) * ldkv + (size_t)g * CHD;
  const __bf16* vhp = v + (size_t)(b * CS) * ldkv + (size_t)g * CHD;

  // Q fragments: rows qb + wid*16 + (lane&15), 4 K-steps of 32 over HD=128
  v16bf qf[4];
  {
    int qrow = qb + wid * 16 + l16;
    const __bf16* qr = qhp + (size_t)qrow * ldq;
#pragma unroll
    for (int ks = 0; ks < 4; ++ks) qf[ks] = load_a_frag(qr + ks * 32, half);
  }

  v8f o_acc[8];
#pragma unroll
  for (int n = 0; n < 8; ++n) o_acc[n] = zero8();
  float m_r[8], l_r[8];
#pragma unroll
  for (int j = 0; j < 8; ++j) { m_r[j] = -1e30f; l_r[j] = 0.f; }

  const float scale = 0.08838834764831845f;  // 1/sqrt(128)
  const int nkb = (qb + BM) / BN;            // causal key-block count (uniform)

  for (int kb = 0; kb < nkb; ++kb) {
#if HAS_TDM
    if (wid == 0) {   // wave-uniform: one wave drives the TDM for the K tile
      tdm_load_tile_bf16((unsigned int)(uintptr_t)&Ks[0][0],
                         khp + (size_t)(kb * BN) * ldkv,
                         /*tensor_d0=*/CHD, /*tensor_d1=*/(unsigned)(CS - kb * BN),
                         /*tile_d0=*/CHD, /*tile_d1=*/BN,
                         /*stride0=*/(unsigned long long)ldkv,
                         /*pad_interval: 64 DWORDs=256B*/ 5u,
                         /*pad_amount:    4 DWORDs= 16B*/ 3u, true);
    }
#endif
    // V tile transpose staging (all threads), overlapped with the TDM
#pragma unroll
    for (int i = 0; i < 4; ++i) {
      int idx = tid + i * 256;     // 0..1023
      int r   = idx >> 4;          // key 0..63
      int c   = (idx & 15) * 8;    // hd 0..120
      size_t goff = (size_t)(kb * BN + r) * ldkv + c;
#if !HAS_TDM
      *(bf16x8*)&Ks[r][c] = *(const bf16x8*)&khp[goff];
#endif
      bf16x8 vv = *(const bf16x8*)&vhp[goff];
      if (kb + 1 < nkb)   // prefetch next V tile (global_prefetch_b8)
        __builtin_prefetch((const void*)(vhp + goff + (size_t)BN * ldkv), 0, 1);
#pragma unroll
      for (int e = 0; e < 8; ++e) Vt[c + e][r] = vv[e];
    }
#if HAS_TDM
    if (wid == 0) __builtin_amdgcn_s_wait_tensorcnt(0);
#endif
    __syncthreads();

    // S = Q * K^T  (B-frag rows = keys, contiguous HD). Batch the fragment
    // loads per k-step so the LDS returns overlap the WMMA chain.
    v8f sacc[4];
#pragma unroll
    for (int t = 0; t < 4; ++t) sacc[t] = zero8();
#pragma unroll
    for (int ks = 0; ks < 4; ++ks) {
      v16bf bk[4];
#pragma unroll
      for (int t = 0; t < 4; ++t)
        bk[t] = load_b_frag(&Ks[t * 16 + l16][ks * 32], half);
#pragma unroll
      for (int t = 0; t < 4; ++t)
        sacc[t] = WMMA_BF16(qf[ks], bk[t], sacc[t]);
    }

    // Online softmax. Row = reg j + 8*half; the row's 16 values live in the
    // same 16-lane half-group -> xor masks 1,2,4,8 stay inside it (wave32).
    const int colbase = kb * BN;
#pragma unroll
    for (int j = 0; j < 8; ++j) {
      int row = qb + wid * 16 + j + 8 * half;
      float mx = -1e30f;
#pragma unroll
      for (int t = 0; t < 4; ++t) {
        int col = colbase + t * 16 + l16;
        float sv = sacc[t][j] * scale;
        sv = (col > row) ? -1e30f : sv;
        sacc[t][j] = sv;
        mx = fmaxf(mx, sv);
      }
      mx = fmaxf(mx, __shfl_xor(mx, 1, 32));
      mx = fmaxf(mx, __shfl_xor(mx, 2, 32));
      mx = fmaxf(mx, __shfl_xor(mx, 4, 32));
      mx = fmaxf(mx, __shfl_xor(mx, 8, 32));
      mx = fmaxf(mx, m_r[j]);
      float alpha = __expf(m_r[j] - mx);
      m_r[j] = mx;
#pragma unroll
      for (int n = 0; n < 8; ++n) o_acc[n][j] *= alpha;
      float rs = 0.f;
#pragma unroll
      for (int t = 0; t < 4; ++t) {
        float pv = __expf(sacc[t][j] - mx);
        sacc[t][j] = pv;
        rs += pv;
      }
      rs += __shfl_xor(rs, 1, 32);
      rs += __shfl_xor(rs, 2, 32);
      rs += __shfl_xor(rs, 4, 32);
      rs += __shfl_xor(rs, 8, 32);
      l_r[j] = l_r[j] * alpha + rs;
    }

    // Stage P (bf16) in this wave's LDS patch, then re-read as A-fragments
#pragma unroll
    for (int t = 0; t < 4; ++t)
#pragma unroll
      for (int j = 0; j < 8; ++j)
        Ps[wid][j + 8 * half][t * 16 + l16] = (__bf16)sacc[t][j];
    asm volatile("s_wait_dscnt 0" ::: "memory");  // wave-internal LDS RAW

    // O += P * V  (B-frag rows = HD cols from Vt, contiguous keys)
#pragma unroll
    for (int ks = 0; ks < 2; ++ks) {
      v16bf af = load_a_frag(&Ps[wid][l16][ks * 32], half);
#pragma unroll
      for (int g2 = 0; g2 < 2; ++g2) {
        v16bf bv[4];
#pragma unroll
        for (int n = 0; n < 4; ++n)
          bv[n] = load_b_frag(&Vt[(g2 * 4 + n) * 16 + l16][ks * 32], half);
#pragma unroll
        for (int n = 0; n < 4; ++n)
          o_acc[g2 * 4 + n] = WMMA_BF16(af, bv[n], o_acc[g2 * 4 + n]);
      }
    }
    __syncthreads();
  }

  // Epilogue: O /= l, write [b][s][h][hd] (== [bs][h*HD+hd])
  __bf16* op = o + (size_t)(b * CS) * ldq + (size_t)h * CHD;
#pragma unroll
  for (int j = 0; j < 8; ++j) {
    float inv = 1.0f / l_r[j];
    int row = qb + wid * 16 + j + 8 * half;
#pragma unroll
    for (int n = 0; n < 8; ++n)
      op[(size_t)row * ldq + n * 16 + l16] = (__bf16)(o_acc[n][j] * inv);
  }
}

// ---------------------------------------------------------------------------
// Host-side launch
// ---------------------------------------------------------------------------
extern "C" void kernel_launch(void* const* d_in, const int* in_sizes, int n_in,
                              void* d_out, int out_size, void* d_ws, size_t ws_size,
                              hipStream_t stream) {
  const float* x  = (const float*)d_in[0];
  const float* fc = (const float*)d_in[1];
  const float* fs = (const float*)d_in[2];
  const float* wq = (const float*)d_in[3];
  const float* wk = (const float*)d_in[4];
  const float* wv = (const float*)d_in[5];
  const float* wo = (const float*)d_in[6];

  // Workspace layout (bf16 elements) — ~192 MiB total
  __bf16* xb  = (__bf16*)d_ws;
  __bf16* wqT = xb  + (size_t)CM * CD;               // [4096][4096]
  __bf16* wkT = wqT + (size_t)CD * CD;               // [1024][4096]
  __bf16* wvT = wkT + (size_t)(CNKV * CHD) * CD;     // [1024][4096]
  __bf16* woT = wvT + (size_t)(CNKV * CHD) * CD;     // [4096][4096]
  __bf16* qb  = woT + (size_t)CD * CD;               // [4096][4096]
  __bf16* kb  = qb  + (size_t)CM * (CNH * CHD);      // [4096][1024]
  __bf16* vb  = kb  + (size_t)CM * (CNKV * CHD);     // [4096][1024]
  __bf16* ao  = vb  + (size_t)CM * (CNKV * CHD);     // [4096][4096]

  // 1) Convert x to bf16
  {
    int n4 = (CM * CD) / 4;
    cvt_bf16_kernel<<<n4 / 256, 256, 0, stream>>>(x, xb, n4);
  }
  // 2) Transpose-convert weights
  cvt_transpose_kernel<<<dim3(CD / 32, CD / 32), 256, 0, stream>>>(wq, wqT, CD, CD);
  cvt_transpose_kernel<<<dim3((CNKV * CHD) / 32, CD / 32), 256, 0, stream>>>(wk, wkT, CD, CNKV * CHD);
  cvt_transpose_kernel<<<dim3((CNKV * CHD) / 32, CD / 32), 256, 0, stream>>>(wv, wvT, CD, CNKV * CHD);
  cvt_transpose_kernel<<<dim3(CD / 32, CD / 32), 256, 0, stream>>>(wo, woT, CNH * CHD, CD);

  // 3) QKV projections (bf16 WMMA GEMMs)
  gemm_bf16_kernel<__bf16><<<dim3(CD / 128, CM / 128), 256, 0, stream>>>(
      xb, wqT, qb, CM, CNH * CHD, CD);
  gemm_bf16_kernel<__bf16><<<dim3((CNKV * CHD) / 128, CM / 128), 256, 0, stream>>>(
      xb, wkT, kb, CM, CNKV * CHD, CD);
  gemm_bf16_kernel<__bf16><<<dim3((CNKV * CHD) / 128, CM / 128), 256, 0, stream>>>(
      xb, wvT, vb, CM, CNKV * CHD, CD);

  // 4) RoPE on Q and K
  rope_kernel<<<CM, 256, 0, stream>>>(qb, fc, fs, CNH);
  rope_kernel<<<CM, 256, 0, stream>>>(kb, fc, fs, CNKV);

  // 5) Causal GQA flash attention
  attn_kernel<<<dim3(CS / 128, CNH, CB), 256, 0, stream>>>(qb, kb, vb, ao);

  // 6) Output projection -> f32 d_out
  gemm_bf16_kernel<float><<<dim3(CD / 128, CM / 128), 256, 0, stream>>>(
      ao, woT, (float*)d_out, CM, CD, CNH * CHD);

  (void)in_sizes; (void)n_in; (void)out_size; (void)ws_size;
}
